// TpcModule_37108517437706
// MI455X (gfx1250) — 8-bit matrix operand order repaired and checked by reading, compiled, not run on hardware
//
#include <hip/hip_runtime.h>
#include <stdint.h>

typedef int   v8i __attribute__((ext_vector_type(8)));
typedef float v8f __attribute__((ext_vector_type(8)));

#define M_DIM 8192
#define K_DIM 64
#define N_DIM 8192
#define M_TILES (M_DIM / 16)
#define N_TILES (N_DIM / 16)
#define PACK_DWORDS (M_TILES * 32 * 8)

__device__ __forceinline__ uint32_t f32_to_e4m3(float x) {
  uint32_t fb = __float_as_uint(x);
  uint32_t sign = fb & 0x80000000u;
  fb ^= sign;
  uint32_t result;
  if (fb >= 0x43F00000u) {
    result = 0x7Eu;
  } else if (fb < (121u << 23)) {
    float f = __uint_as_float(fb) + __uint_as_float(141u << 23);
    result = __float_as_uint(f) - (141u << 23);
  } else {
    uint32_t mant_odd = (fb >> 20) & 1u;
    fb += 0xC4000000u + 0x7FFFFu;
    fb += mant_odd;
    result = fb >> 20;
  }
  return result | (sign >> 24);
}

__global__ void __launch_bounds__(256) pack_a_kernel(const float* __restrict__ q,
                                                     uint32_t* __restrict__ apack) {
  int t = blockIdx.x * blockDim.x + threadIdx.x;
  int j    = t & 7;
  int lane = (t >> 3) & 31;
  int tile = t >> 8;
  int row  = tile * 16 + (lane & 15);
  int half = lane >> 4;
  int kb   = ((j >> 1) << 4) + ((j & 1) << 2) + (half << 3);
  const float* src = q + row * K_DIM + kb;
  uint32_t b0 = f32_to_e4m3(src[0]);
  uint32_t b1 = f32_to_e4m3(src[1]);
  uint32_t b2 = f32_to_e4m3(src[2]);
  uint32_t b3 = f32_to_e4m3(src[3]);
  apack[t] = b0 | (b1 << 8) | (b2 << 16) | (b3 << 24);
}

__global__ void __launch_bounds__(256) pack_b_kernel(const float* __restrict__ kmat,
                                                     uint32_t* __restrict__ bpack) {
  int t = blockIdx.x * blockDim.x + threadIdx.x;
  int j    = t & 7;
  int lane = (t >> 3) & 31;
  int tile = t >> 8;
  int col  = tile * 16 + (lane & 15);
  int kb   = ((j >> 1) << 4) + ((lane >> 4) << 3) + ((j & 1) << 2);
  const float* src = kmat + (long)kb * N_DIM + col;
  uint32_t b0 = f32_to_e4m3(src[0 * N_DIM]);
  uint32_t b1 = f32_to_e4m3(src[1 * N_DIM]);
  uint32_t b2 = f32_to_e4m3(src[2 * N_DIM]);
  uint32_t b3 = f32_to_e4m3(src[3 * N_DIM]);
  bpack[t] = b0 | (b1 << 8) | (b2 << 16) | (b3 << 24);
}

__global__ void __launch_bounds__(256) gemm_fp8_kernel(const v8i* __restrict__ apack,
                                                       const v8i* __restrict__ bpack,
                                                       float* __restrict__ out) {
  const int lane = threadIdx.x & 31;
  const int wave = blockIdx.x * (blockDim.x >> 5) + (threadIdx.x >> 5);
  const int r = wave >> 7;
  const int c = wave & 127;

  v8i a = apack[r * 32 + lane];

  const int mBase = r * 16 + ((lane >> 4) << 3);
  const int nLane = lane & 15;
  float* outBase = out + (long)mBase * N_DIM + nLane;

#pragma unroll
  for (int t = 0; t < 4; ++t) {
    const int nTile = c * 4 + t;
    v8i b = bpack[nTile * 32 + lane];
    v8f acc = {};
    acc = __builtin_amdgcn_wmma_f32_16x16x64_fp8_fp8(a, b, (short)0, acc,
                                                      false,
                                                      false);
    float* o = outBase + nTile * 16;
#pragma unroll
    for (int v = 0; v < 8; ++v) {
      o[(long)v * N_DIM] = acc[v];
    }
  }
}

extern "C" void kernel_launch(void* const* d_in, const int* in_sizes, int n_in,
                              void* d_out, int out_size, void* d_ws, size_t ws_size,
                              hipStream_t stream) {
  const float* q = (const float*)d_in[0];
  const float* k = (const float*)d_in[1];
  float* out = (float*)d_out;

  uint32_t* apack = (uint32_t*)d_ws;
  uint32_t* bpack = apack + PACK_DWORDS;

  pack_a_kernel<<<PACK_DWORDS / 256, 256, 0, stream>>>(q, apack);
  pack_b_kernel<<<PACK_DWORDS / 256, 256, 0, stream>>>(k, bpack);

  gemm_fp8_kernel<<<65536 / 8, 256, 0, stream>>>((const v8i*)apack,
                                                 (const v8i*)bpack, out);
}
